// attention_action_decoder_20727512170626
// MI455X (gfx1250) — matrix operand order repaired and checked by reading, not compiled, not run on hardware
//
#include <hip/hip_runtime.h>
#include <hip/hip_bf16.h>
#include <math.h>

typedef __attribute__((ext_vector_type(16))) __bf16 v16bf;
typedef __attribute__((ext_vector_type(8)))  float  v8f;

__device__ __forceinline__ float sigmoidf_(float x) { return 1.f / (1.f + __expf(-x)); }

// ---------------------------------------------------------------------------
// WMMA fragment loaders (bf16, 16x16x32). A: 16x32 tile, row-major with ld.
// Lanes 0-15 hold M=lane, K 0..15; lanes 16-31 hold K 16..31.
// B: weight tile stored row-major [N][K]; B[k][n] = W[n][k], lanes 0-15 -> N=lane,
// K 0..15; lanes 16-31 -> K 16..31 (one contiguous 32B read per lane).
// ---------------------------------------------------------------------------
__device__ __forceinline__ v16bf load_a_frag(const __bf16* A, int lda, int lane) {
  const int m  = lane & 15;
  const int kb = (lane >> 4) << 4; // 0 or 16
  const __bf16* p = A + (size_t)m * lda + kb;
  v16bf f;
#pragma unroll
  for (int i = 0; i < 8; ++i) f[i] = p[i];
#pragma unroll
  for (int i = 0; i < 8; ++i) f[8 + i] = p[8 + i];
  return f;
}

__device__ __forceinline__ v16bf load_b_frag(const __bf16* W, int ldw, int lane) {
  const int n  = lane & 15;
  const int kb = (lane >> 4) << 4; // 0 or 16
  const __bf16* p = W + (size_t)n * ldw + kb;
  v16bf f;
#pragma unroll
  for (int i = 0; i < 16; ++i) f[i] = p[i];
  return f;
}

// D(16xN) = [A0(16,k0) | A1(16,k1)] @ W^T,  W row-major [N][k0+k1] bf16.
// 8 waves split the N tiles, two tiles per pass so each A fragment feeds two
// v_wmma issues (halves LDS traffic per WMMA). Out is f32 or bf16 LDS, [16][ldo].
template <typename OutT>
__device__ __forceinline__ void wmma_gemm2(const __bf16* A0, int k0,
                                           const __bf16* A1, int k1, int lda,
                                           const __bf16* W, int N,
                                           OutT* out, int ldo, int tid) {
  const int lane = tid & 31, wave = tid >> 5;
  const int K = k0 + k1;
  const int T = N >> 4;
  for (int nt = wave << 1; nt < T; nt += 16) {
    const int nc0 = nt << 4, nc1 = nc0 + 16;
    const __bf16* W0 = W + (size_t)nc0 * K;
    const __bf16* W1 = W + (size_t)nc1 * K;
    // warm the next pair of weight rows (gfx1250 global_prefetch_b8)
    __builtin_prefetch(W0 + (size_t)32 * K, 0, 1);
    v8f acc0 = {}, acc1 = {};
    for (int k = 0; k < K; k += 32) {
      const __bf16* Ap = (k < k0) ? (A0 + k) : (A1 + (k - k0));
      v16bf a  = load_a_frag(Ap, lda, lane);
      v16bf b0 = load_b_frag(W0 + k, K, lane);
      v16bf b1 = load_b_frag(W1 + k, K, lane);
      acc0 = __builtin_amdgcn_wmma_f32_16x16x32_bf16(false, a, false, b0,
                                                     (short)0, acc0, false, false);
      acc1 = __builtin_amdgcn_wmma_f32_16x16x32_bf16(false, a, false, b1,
                                                     (short)0, acc1, false, false);
    }
    const int n = lane & 15, mb = (lane >> 4) << 3;
#pragma unroll
    for (int v = 0; v < 8; ++v) {
      out[(size_t)(mb + v) * ldo + nc0 + n] = (OutT)acc0[v];
      out[(size_t)(mb + v) * ldo + nc1 + n] = (OutT)acc1[v];
    }
  }
}

// ---------------------------------------------------------------------------
// Batched attention over Hm[B][256][512] (f32, L2-resident) for the 16-row
// batch tile: scores = Hm . u ; softmax ; z = alpha . Hm -> bf16 into Abuf.
// ---------------------------------------------------------------------------
__device__ void attend_block(const float* __restrict__ Hm, int b0,
                             const float* ubuf, float* scores,
                             __bf16* Abuf, int zoff, int tid) {
  // scores[b][l]   (thread: l = tid, b = i)
  for (int i = 0; i < 16; ++i) {
    int p = tid + (i << 8);
    int b = p >> 8, l = p & 255;
    const float4* row = (const float4*)(Hm + (((size_t)(b0 + b)) * 256 + l) * 512);
    const float4* uq  = (const float4*)(ubuf + (size_t)b * 512);
    float s = 0.f;
#pragma unroll 4
    for (int k = 0; k < 128; ++k) {
      float4 a = row[k], q = uq[k];
      s = fmaf(a.x, q.x, fmaf(a.y, q.y, fmaf(a.z, q.z, fmaf(a.w, q.w, s))));
    }
    scores[(b << 8) + l] = s;
  }
  __syncthreads();
  // softmax per batch row
  if (tid < 16) {
    float* s = scores + (tid << 8);
    float mx = -3.0e38f;
    for (int l = 0; l < 256; ++l) mx = fmaxf(mx, s[l]);
    float sum = 0.f;
    for (int l = 0; l < 256; ++l) { float e = __expf(s[l] - mx); s[l] = e; sum += e; }
    float inv = 1.f / sum;
    for (int l = 0; l < 256; ++l) s[l] *= inv;
  }
  __syncthreads();
  // z[b][d] = sum_l alpha[b][l] * Hm[b][l][d]  (float4 over d, coalesced in d)
  for (int i = 0; i < 8; ++i) {
    int p = tid + (i << 8);
    int b = p >> 7, d4 = p & 127;
    const float4* col = (const float4*)(Hm + ((size_t)(b0 + b)) * 256 * 512) + d4;
    const float* al = scores + (b << 8);
    float4 acc = make_float4(0.f, 0.f, 0.f, 0.f);
    for (int l = 0; l < 256; ++l) {
      float4 v = col[(size_t)l * 128];
      float a = al[l];
      acc.x = fmaf(a, v.x, acc.x); acc.y = fmaf(a, v.y, acc.y);
      acc.z = fmaf(a, v.z, acc.z); acc.w = fmaf(a, v.w, acc.w);
    }
    __bf16* dst = Abuf + (size_t)b * 1536 + zoff + (d4 << 2);
    dst[0] = (__bf16)acc.x; dst[1] = (__bf16)acc.y;
    dst[2] = (__bf16)acc.z; dst[3] = (__bf16)acc.w;
  }
}

// ---------------------------------------------------------------------------
// Prep kernels: f32 -> bf16 weight conversion / fusion / constant folding
// ---------------------------------------------------------------------------
__global__ void cvt_bf16_2d(const float* __restrict__ src, int src_ld, int src_rows,
                            __bf16* __restrict__ dst, int rows, int cols) {
  size_t n = (size_t)rows * cols;
  for (size_t i = (size_t)blockIdx.x * blockDim.x + threadIdx.x; i < n;
       i += (size_t)gridDim.x * blockDim.x) {
    int r = (int)(i / cols), c = (int)(i % cols);
    float v = (r < src_rows) ? src[(size_t)r * src_ld + c] : 0.f;
    dst[i] = (__bf16)v;
  }
}

// Wcomb[u*4+g][k] = (k<512 ? W_ih : W_hh)[g*512+u][k mod 512]  (gate order i,f,g,o)
__global__ void build_wcomb(const float* __restrict__ Wih, const float* __restrict__ Whh,
                            __bf16* __restrict__ dst) {
  size_t n = (size_t)2048 * 1024;
  for (size_t i = (size_t)blockIdx.x * blockDim.x + threadIdx.x; i < n;
       i += (size_t)gridDim.x * blockDim.x) {
    int q = (int)(i >> 10), k = (int)(i & 1023);
    int u = q >> 2, g = q & 3;
    int row = g * 512 + u;
    float v = (k < 512) ? Wih[(size_t)row * 512 + k] : Whh[(size_t)row * 512 + (k - 512)];
    dst[i] = (__bf16)v;
  }
}

// Degenerate attentions fold to the context vectors -> constant per batch row:
// pre[b][j] = Wbd_b[j] + sum_k (Wbd[j][1024+k]+Wbd[j][1152+k])*ini[b][k]
//                      + sum_k (Wbd[j][1280+k]+Wbd[j][1408+k])*cur[b][k]
__global__ void prep_pre(const float* __restrict__ Wbd, const float* __restrict__ Wbd_b,
                         const float* __restrict__ ini, const float* __restrict__ cur,
                         float* __restrict__ pre) {
  int b = blockIdx.x, j = threadIdx.x;
  const float* wr = Wbd + (size_t)j * 1664 + 1024;
  const float* iv = ini + (size_t)b * 128;
  const float* cv = cur + (size_t)b * 128;
  float s = Wbd_b[j];
  for (int k = 0; k < 128; ++k) {
    s = fmaf(wr[k] + wr[128 + k], iv[k], s);
    s = fmaf(wr[256 + k] + wr[384 + k], cv[k], s);
  }
  pre[(size_t)b * 512 + j] = s;
}

// xpart[b*32+t][j] = Wbd[j][1536:1664] . emb[actions[b][t]]
__global__ void prep_xpart(const float* __restrict__ Wbd, const int* __restrict__ actions,
                           const float* __restrict__ emb, float* __restrict__ xp) {
  int bt = blockIdx.x, j = threadIdx.x;
  int a = actions[bt];
  const float* e  = emb + (size_t)a * 128;
  const float* wr = Wbd + (size_t)j * 1664 + 1536;
  float s = 0.f;
  for (int k = 0; k < 128; ++k) s = fmaf(wr[k], e[k], s);
  xp[(size_t)bt * 512 + j] = s;
}

// ---------------------------------------------------------------------------
// Main recurrence: 8 workgroups x 16 batch rows, 32 sequential steps.
// ---------------------------------------------------------------------------
__global__ __launch_bounds__(256) void decoder_main(
    const float* __restrict__ ins, const float* __restrict__ his,
    const float* __restrict__ pre, const float* __restrict__ xpart,
    const float* __restrict__ b_ih, const float* __restrict__ b_hh,
    const float* __restrict__ h0, const float* __restrict__ c0,
    const __bf16* __restrict__ wWc, const __bf16* __restrict__ wWp,
    const __bf16* __restrict__ wWbd, const __bf16* __restrict__ wWcomb,
    __bf16* __restrict__ outsb) {
  __shared__ __align__(16) unsigned char smem[155648]; // 152 KB
  __bf16* Abuf   = (__bf16*)smem;                // 16 x 1536 bf16: [h | z_c | z_p->hk]
  float*  cbuf   = (float*)(smem + 49152);       // 16 x 512 f32 cell state
  float*  ubuf   = (float*)(smem + 81920);       // 16 x 512 f32 (attention query)
  float*  scores = (float*)(smem + 114688);      // 16 x 256 f32
  __bf16* gatesb = (__bf16*)(smem + 81920);      // 16 x 2048 bf16 (overlays ubuf+scores)
  float*  biasg  = (float*)(smem + 147456);      // 2048 f32, interleaved [unit][gate]

  const int tid = threadIdx.x;
  const int b0  = blockIdx.x << 4;

  // init h (bf16), c (f32), interleaved LSTM bias
  for (int i = 0; i < 32; ++i) {
    int p = tid + (i << 8);
    int b = p >> 9, j = p & 511;
    Abuf[(size_t)b * 1536 + j] = (__bf16)h0[((size_t)(b0 + b)) * 512 + j];
    cbuf[(b << 9) + j] = c0[((size_t)(b0 + b)) * 512 + j];
  }
  for (int i = 0; i < 8; ++i) {
    int q = tid + (i << 8);            // q = u*4 + g
    int u = q >> 2, g = q & 3;
    biasg[q] = b_ih[g * 512 + u] + b_hh[g * 512 + u];
  }
  __syncthreads();

  for (int t = 0; t < 32; ++t) {
    // (1) u = h @ W_c^T              (16x512x512)
    wmma_gemm2<float>(Abuf, 512, Abuf, 0, 1536, wWc, 512, ubuf, 512, tid);
    __syncthreads();
    // (2) z_c = attend(ins, u) -> Abuf slot 512
    attend_block(ins, b0, ubuf, scores, Abuf, 512, tid);
    __syncthreads();
    // (3) u_p = [h,z_c] @ W_p^T      (16x1024x512, contiguous A)
    wmma_gemm2<float>(Abuf, 1024, Abuf, 0, 1536, wWp, 512, ubuf, 512, tid);
    __syncthreads();
    // (4) z_p = attend(his, u_p) -> Abuf slot 1024
    attend_block(his, b0, ubuf, scores, Abuf, 1024, tid);
    __syncthreads();
    // (5) hk_pre = [z_c,z_p] @ Wbd[:, :1024]^T  (contiguous A at +512)
    wmma_gemm2<float>(Abuf + 512, 1024, Abuf, 0, 1536, wWbd, 512, ubuf, 512, tid);
    __syncthreads();
    // (6) hk = tanh(hk_pre + pre + xpart) -> slot 1024 (overwrites z_p)
    for (int i = 0; i < 32; ++i) {
      int p = tid + (i << 8);
      int b = p >> 9, j = p & 511;
      float v = ubuf[(b << 9) + j] + pre[((size_t)(b0 + b)) * 512 + j] +
                xpart[(((size_t)(b0 + b)) * 32 + t) * 512 + j];
      Abuf[(size_t)b * 1536 + 1024 + j] = (__bf16)tanhf(v);
    }
    __syncthreads();
    // (7) gates = [hk,h] @ Wcomb^T   (16x1024x2048, i/f/g/o interleaved per unit)
    wmma_gemm2<__bf16>(Abuf + 1024, 512, Abuf, 512, 1536, wWcomb, 2048, gatesb, 2048, tid);
    __syncthreads();
    // (8) LSTM elementwise update
    for (int i = 0; i < 32; ++i) {
      int p = tid + (i << 8);
      int b = p >> 9, u = p & 511;
      const __bf16* g = gatesb + (b << 11) + (u << 2);
      const float* bb = biasg + (u << 2);
      float gi = (float)g[0] + bb[0];
      float gf = (float)g[1] + bb[1];
      float gg = (float)g[2] + bb[2];
      float go = (float)g[3] + bb[3];
      float cn = sigmoidf_(gf) * cbuf[(b << 9) + u] + sigmoidf_(gi) * tanhf(gg);
      float hn = sigmoidf_(go) * tanhf(cn);
      cbuf[(b << 9) + u] = cn;
      __bf16 hb = (__bf16)hn;
      Abuf[(size_t)b * 1536 + u] = hb;
      outsb[(((size_t)(b0 + b)) * 32 + t) * 512 + u] = hb;
    }
    __syncthreads();
  }
}

// ---------------------------------------------------------------------------
// Final projection: out[4096][1000] = outs(bf16) @ Wha^T + b. Wide parallel WMMA.
// ---------------------------------------------------------------------------
__global__ __launch_bounds__(256) void proj_kernel(const __bf16* __restrict__ outsb,
                                                   const __bf16* __restrict__ WhaB,
                                                   const float* __restrict__ bias,
                                                   float* __restrict__ out) {
  const int tid = threadIdx.x, lane = tid & 31, wave = tid >> 5;
  const int mt   = blockIdx.x << 4;
  const int ncol = (blockIdx.y << 7) + (wave << 4);
  const __bf16* A  = outsb + (size_t)mt * 512;
  const __bf16* W0 = WhaB + (size_t)ncol * 512;
  __builtin_prefetch(W0, 0, 1);
  v8f acc = {};
  for (int k = 0; k < 512; k += 32) {
    v16bf a = load_a_frag(A + k, 512, lane);
    v16bf b = load_b_frag(W0 + k, 512, lane);
    acc = __builtin_amdgcn_wmma_f32_16x16x32_bf16(false, a, false, b,
                                                  (short)0, acc, false, false);
  }
  const int n = lane & 15, mb = (lane >> 4) << 3;
  const int col = ncol + n;
  if (col < 1000) {
    float bv = bias[col];
#pragma unroll
    for (int v = 0; v < 8; ++v)
      out[(size_t)(mt + mb + v) * 1000 + col] = acc[v] + bv;
  }
}

// ---------------------------------------------------------------------------
extern "C" void kernel_launch(void* const* d_in, const int* in_sizes, int n_in,
                              void* d_out, int out_size, void* d_ws, size_t ws_size,
                              hipStream_t stream) {
  (void)in_sizes; (void)n_in; (void)out_size; (void)ws_size;
  const float* ins     = (const float*)d_in[0];
  const float* his     = (const float*)d_in[1];
  const int*   actions = (const int*)d_in[2];
  const float* cur     = (const float*)d_in[3];
  const float* ini     = (const float*)d_in[4];
  const float* emb     = (const float*)d_in[5];
  const float* W_c     = (const float*)d_in[6];
  const float* W_p     = (const float*)d_in[7];
  // d_in[8..11] = W_s_* : unused (degenerate attentions fold away)
  const float* Wbd_w   = (const float*)d_in[12];
  const float* Wbd_b   = (const float*)d_in[13];
  const float* W_ih    = (const float*)d_in[14];
  const float* W_hh    = (const float*)d_in[15];
  const float* b_ih    = (const float*)d_in[16];
  const float* b_hh    = (const float*)d_in[17];
  const float* Wha_w   = (const float*)d_in[18];
  const float* Wha_b   = (const float*)d_in[19];
  const float* h0      = (const float*)d_in[20];
  const float* c0      = (const float*)d_in[21];

  char* ws = (char*)d_ws;
  __bf16* wWc    = (__bf16*)(ws + 0);         //  512x512
  __bf16* wWp    = (__bf16*)(ws + 524288);    //  512x1024
  __bf16* wWbd   = (__bf16*)(ws + 1572864);   //  512x1024 (cols 0..1023 of Wbd)
  __bf16* wWcomb = (__bf16*)(ws + 2621440);   // 2048x1024
  __bf16* wWha   = (__bf16*)(ws + 6815744);   // 1024x512 (rows >=1000 zero)
  float*  pre    = (float*)(ws + 7864320);    //  128x512
  float*  xpart  = (float*)(ws + 8126464);    // 4096x512
  __bf16* outsb  = (__bf16*)(ws + 16515072);  // 4096x512

  cvt_bf16_2d<<<512, 256, 0, stream>>>(W_c, 512, 512, wWc, 512, 512);
  cvt_bf16_2d<<<1024, 256, 0, stream>>>(W_p, 1024, 512, wWp, 512, 1024);
  cvt_bf16_2d<<<1024, 256, 0, stream>>>(Wbd_w, 1664, 512, wWbd, 512, 1024);
  cvt_bf16_2d<<<1024, 256, 0, stream>>>(Wha_w, 512, 1000, wWha, 1024, 512);
  build_wcomb<<<2048, 256, 0, stream>>>(W_ih, W_hh, wWcomb);
  prep_pre<<<128, 512, 0, stream>>>(Wbd_w, Wbd_b, ini, cur, pre);
  prep_xpart<<<4096, 512, 0, stream>>>(Wbd_w, actions, emb, xpart);

  decoder_main<<<8, 256, 0, stream>>>(ins, his, pre, xpart, b_ih, b_hh, h0, c0,
                                      wWc, wWp, wWbd, wWcomb, outsb);

  proj_kernel<<<dim3(256, 8), 256, 0, stream>>>(outsb, wWha, Wha_b, (float*)d_out);
}
